// CustomLossFunction_58007828300479
// MI455X (gfx1250) — compile-verified, matrix operand order checked by reading, not run on hardware
//
#include <hip/hip_runtime.h>
#include <hip/hip_bf16.h>

typedef __attribute__((ext_vector_type(16))) _Float16 v16h;
typedef __attribute__((ext_vector_type(8)))  _Float16 v8h;
typedef __attribute__((ext_vector_type(4)))  _Float16 v4h;
typedef __attribute__((ext_vector_type(8)))  float    v8f;
typedef __attribute__((ext_vector_type(4)))  float    v4f;

#define B_ 16
#define N_ 1024
#define D_ 128
#define C_ 2

// workspace float offsets
#define WS_BCE   0
#define WS_COORD 1
#define WS_REG   2
#define WS_POS_S 16
#define WS_NEG_S 32
#define WS_POS_C 48
#define WS_NEG_C 64
#define WS_CNT   80
#define WS_NRM   256
#define WS_FH_BYTES ((WS_NRM + B_ * N_) * 4)   // byte offset of f16 feature buffer

__device__ __forceinline__ float wave_red(float v) {
#pragma unroll
  for (int off = 16; off > 0; off >>= 1) v += __shfl_xor(v, off, 32);
  return v;
}

// -------------------------------------------------------------------------
// Prep: f32 -> f16 features, per-row squared norms (f32), reg/coord/count
// partial sums into ws accumulators.
// -------------------------------------------------------------------------
__global__ __launch_bounds__(256) void prep_kernel(
    const float* __restrict__ feat, const float* __restrict__ pcoords,
    const float* __restrict__ points, const float* __restrict__ masks,
    float* __restrict__ ws, _Float16* __restrict__ fh) {
  const int lane   = threadIdx.x & 31;
  const int wave   = (blockIdx.x * blockDim.x + threadIdx.x) >> 5;
  const int nwaves = (gridDim.x * blockDim.x) >> 5;
  const int tid    = blockIdx.x * blockDim.x + threadIdx.x;
  const int nth    = gridDim.x * blockDim.x;

  // ---- features: one row (128 f32) per wave; convert + squared norm ----
  float regsum = 0.f;
  for (int row = wave; row < B_ * N_; row += nwaves) {
    v4f f = *(const v4f*)(feat + (size_t)row * D_ + lane * 4);
    v4h h;
#pragma unroll
    for (int e = 0; e < 4; e++) h[e] = (_Float16)f[e];
    *(v4h*)(fh + (size_t)row * D_ + lane * 4) = h;
    float s = f[0] * f[0] + f[1] * f[1] + f[2] * f[2] + f[3] * f[3];
    s = wave_red(s);
    if (lane == 0) { ws[WS_NRM + row] = s; regsum += s; }
  }
  if (lane == 0) atomicAdd(ws + WS_REG, regsum);

  // ---- coord MSE partial ----
  float csum = 0.f;
  for (int i = tid; i < B_ * N_ * C_; i += nth) {
    float d = pcoords[i] - points[i];
    csum += d * d;
  }
  csum = wave_red(csum);
  if (lane == 0) atomicAdd(ws + WS_COORD, csum);

  // ---- per-batch node-mask counts (32-elem chunks, 32 chunks per batch) ----
  for (int chunk = wave; chunk < (B_ * N_) / 32; chunk += nwaves) {
    float v = masks[chunk * 32 + lane];
    float s = wave_red(v);
    if (lane == 0) atomicAdd(ws + WS_CNT + (chunk >> 5), s);
  }
}

// -------------------------------------------------------------------------
// Fragment loader: 16-bit A-layout (B is its mirror, same load works since
// B = F_j^T). lane<16: K in {k0..k0+7, k0+16..k0+23}; lane>=16: +8.
// -------------------------------------------------------------------------
__device__ __forceinline__ v16h load_frag(const _Float16* __restrict__ fb,
                                          int row0, int k0, int lane) {
  const _Float16* p = fb + (size_t)(row0 + (lane & 15)) * D_ + k0 + ((lane >> 4) << 3);
  v8h lo = *(const v8h*)p;
  v8h hi = *(const v8h*)(p + 16);
  v16h r;
#pragma unroll
  for (int e = 0; e < 8; e++) { r[e] = lo[e]; r[e + 8] = hi[e]; }
  return r;
}

// -------------------------------------------------------------------------
// Fused Gram (WMMA f16->f32) + BCE + contrastive pos/neg over one 64x128
// tile of the (i,j) pair space. 8 waves, each owns a 32x32 region.
// Streaming tensors (logits/adj) use non-temporal loads so the f16 feature
// buffer (4 MB) stays resident in the 192 MB L2 for fragment re-reads.
// -------------------------------------------------------------------------
__global__ __launch_bounds__(256) void gram_edge_kernel(
    const _Float16* __restrict__ fh, const float* __restrict__ nrm,
    const float* __restrict__ logits, const float* __restrict__ adj,
    float* __restrict__ ws) {
  const int b      = blockIdx.z;
  const int i0     = blockIdx.y * 64;
  const int j0     = blockIdx.x * 128;
  const int lane   = threadIdx.x & 31;
  const int waveId = threadIdx.x >> 5;
  const int wm = waveId >> 2, wn = waveId & 3;
  const int ib = i0 + wm * 32, jb = j0 + wn * 32;
  const _Float16* fb = fh + (size_t)b * N_ * D_;

  const v8f vzero = {0.f, 0.f, 0.f, 0.f, 0.f, 0.f, 0.f, 0.f};
  v8f acc[2][2];
#pragma unroll
  for (int a = 0; a < 2; a++)
#pragma unroll
    for (int c = 0; c < 2; c++) acc[a][c] = vzero;

#pragma unroll
  for (int k0 = 0; k0 < D_; k0 += 32) {
    v16h a0  = load_frag(fb, ib,      k0, lane);
    v16h a1  = load_frag(fb, ib + 16, k0, lane);
    v16h bf0 = load_frag(fb, jb,      k0, lane);
    v16h bf1 = load_frag(fb, jb + 16, k0, lane);
    acc[0][0] = __builtin_amdgcn_wmma_f32_16x16x32_f16(false, a0, false, bf0,
                                                       (short)0, acc[0][0], false, false);
    acc[0][1] = __builtin_amdgcn_wmma_f32_16x16x32_f16(false, a0, false, bf1,
                                                       (short)0, acc[0][1], false, false);
    acc[1][0] = __builtin_amdgcn_wmma_f32_16x16x32_f16(false, a1, false, bf0,
                                                       (short)0, acc[1][0], false, false);
    acc[1][1] = __builtin_amdgcn_wmma_f32_16x16x32_f16(false, a1, false, bf1,
                                                       (short)0, acc[1][1], false, false);
  }

  // Epilogue: C/D layout -> element (M = v + 8*(lane>>4), N = lane&15)
  float bce = 0.f, ps = 0.f, ns = 0.f, pc = 0.f, nc = 0.f;
  const size_t base = (size_t)b * N_ * N_;
  const int h = lane >> 4, nn = lane & 15;
#pragma unroll
  for (int tm = 0; tm < 2; tm++) {
#pragma unroll
    for (int tn = 0; tn < 2; tn++) {
      const int jj  = jb + tn * 16 + nn;
      const float nj = nrm[b * N_ + jj];
#pragma unroll
      for (int v = 0; v < 8; v++) {
        const int ii   = ib + tm * 16 + v + 8 * h;
        const float ni = nrm[b * N_ + ii];
        const float g  = acc[tm][tn][v];
        const size_t idx = base + (size_t)ii * N_ + jj;
        const float l = __builtin_nontemporal_load(logits + idx);
        const float a = __builtin_nontemporal_load(adj + idx);
        // stable BCE: max(l,0) - l*a + log(1 + exp(-|l|)); x = exp(-|l|) is in
        // (0,1] so plain log(1+x) matches log1p to f32 precision here.
        bce += fmaxf(l, 0.f) - l * a + __logf(1.f + __expf(-fabsf(l)));
        float sq = fmaxf(ni + nj - 2.f * g, 0.f);
        if (a > 0.5f) {
          ps += sq; pc += 1.f;
        } else if (ii != jj) {
          float d  = sqrtf(fmaxf(sq, 1e-12f));
          float hv = fmaxf(1.f - d, 0.f);
          ns += hv * hv; nc += 1.f;
        }
      }
    }
  }

  __shared__ float sred[5];
  if (threadIdx.x < 5) sred[threadIdx.x] = 0.f;
  __syncthreads();
  bce = wave_red(bce); ps = wave_red(ps); ns = wave_red(ns);
  pc  = wave_red(pc);  nc = wave_red(nc);
  if (lane == 0) {
    atomicAdd(&sred[0], bce); atomicAdd(&sred[1], ps); atomicAdd(&sred[2], ns);
    atomicAdd(&sred[3], pc);  atomicAdd(&sred[4], nc);
  }
  __syncthreads();
  if (threadIdx.x == 0) {
    atomicAdd(ws + WS_BCE,       sred[0]);
    atomicAdd(ws + WS_POS_S + b, sred[1]);
    atomicAdd(ws + WS_NEG_S + b, sred[2]);
    atomicAdd(ws + WS_POS_C + b, sred[3]);
    atomicAdd(ws + WS_NEG_C + b, sred[4]);
  }
}

// -------------------------------------------------------------------------
// Finalize: one wave assembles the 6 outputs.
// -------------------------------------------------------------------------
__global__ void finalize_kernel(const float* __restrict__ ws,
                                const float* __restrict__ pcount,
                                float* __restrict__ out) {
  const int t = threadIdx.x;  // 32 threads
  float cl = 0.f, ct = 0.f;
  if (t < B_) {
    float cnt = ws[WS_CNT + t];
    float d = fabsf(pcount[t] - cnt);
    cl = (d < 1.f) ? 0.5f * d * d : d - 0.5f;
    float p = ws[WS_POS_S + t] / fmaxf(ws[WS_POS_C + t], 1.f);
    float n = ws[WS_NEG_S + t] / fmaxf(ws[WS_NEG_C + t], 1.f);
    ct = p + n;
  }
  cl = wave_red(cl) / (float)B_;
  ct = wave_red(ct) / (float)B_;
  if (t == 0) {
    float coord = ws[WS_COORD] / (float)(B_ * N_ * C_);
    float edge  = ws[WS_BCE]   / ((float)B_ * (float)N_ * (float)N_);
    float reg   = ws[WS_REG]   / (float)(B_ * N_ * D_);
    float total = 1.0f * coord + 2.0f * edge + 0.1f * cl + 0.001f * reg + 0.1f * ct;
    out[0] = total; out[1] = coord; out[2] = edge;
    out[3] = cl;    out[4] = reg;   out[5] = ct;
  }
}

extern "C" void kernel_launch(void* const* d_in, const int* in_sizes, int n_in,
                              void* d_out, int out_size, void* d_ws, size_t ws_size,
                              hipStream_t stream) {
  const float* pcoords = (const float*)d_in[0];
  const float* points  = (const float*)d_in[1];
  const float* logits  = (const float*)d_in[2];
  const float* adj     = (const float*)d_in[3];
  const float* masks   = (const float*)d_in[4];
  const float* pcount  = (const float*)d_in[5];
  const float* feat    = (const float*)d_in[6];
  float*     ws = (float*)d_ws;
  _Float16*  fh = (_Float16*)((char*)d_ws + WS_FH_BYTES);

  // zero scalar/per-batch accumulators (first 256 floats)
  hipMemsetAsync(d_ws, 0, 1024, stream);

  prep_kernel<<<128, 256, 0, stream>>>(feat, pcoords, points, masks, ws, fh);

  // grid: x = j-tile (128 cols -> 8), y = i-tile (64 rows -> 16), z = batch
  gram_edge_kernel<<<dim3(8, 16, 16), 256, 0, stream>>>(fh, ws + WS_NRM,
                                                        logits, adj, ws);

  finalize_kernel<<<1, 32, 0, stream>>>(ws, pcount, (float*)d_out);
}
